// EncoderLayer_42116449305233
// MI455X (gfx1250) — matrix operand order repaired and checked by reading, not compiled, not run on hardware
//
#include <hip/hip_runtime.h>
#include <hip/hip_bf16.h>

typedef __attribute__((ext_vector_type(16))) __bf16 v16bf;
typedef __attribute__((ext_vector_type(8)))  float  v8f;
typedef __attribute__((ext_vector_type(4)))  int    i4;

union Frag {
    uint4 u4[2];
    v16bf v;
};

__device__ __forceinline__ unsigned short f2bf(float f) {
    unsigned u = __float_as_uint(f);
    u += 0x7FFFu + ((u >> 16) & 1u);   // round-to-nearest-even
    return (unsigned short)(u >> 16);
}

__device__ __forceinline__ v8f vzero8() {
    v8f r;
#pragma unroll
    for (int i = 0; i < 8; ++i) r[i] = 0.0f;
    return r;
}

__device__ __forceinline__ v8f wmma_bf16(const Frag& a, const Frag& b, v8f c) {
    return __builtin_amdgcn_wmma_f32_16x16x32_bf16(
        /*neg_a=*/false, a.v, /*neg_b=*/false, b.v,
        /*c_mod=*/(short)0, c, /*reuse_a=*/false, /*reuse_b=*/false);
}

// ---- CDNA5 async global->LDS copy (ASYNCcnt path), with sync fallback ----
#if __has_builtin(__builtin_amdgcn_global_load_async_to_lds_b128) && \
    __has_builtin(__builtin_amdgcn_s_wait_asynccnt)
#define HAVE_ASYNC 1
__device__ __forceinline__ void cp16_async(void* lds, const void* g) {
    __builtin_amdgcn_global_load_async_to_lds_b128(
        (i4*)g, (i4*)lds, /*offset=*/0, /*cpol=*/0);
}
#define ASYNC_WAIT() __builtin_amdgcn_s_wait_asynccnt(0)
#else
#define HAVE_ASYNC 0
__device__ __forceinline__ void cp16_async(void* lds, const void* g) {
    *(uint4*)lds = *(const uint4*)g;
}
#define ASYNC_WAIT() ((void)0)
#endif

// ---------------------------------------------------------------- constants
#define SEQ   2048
#define DM    768
#define NH    12
#define DKH   64
#define HID   3072
#define ROWS  4096   // B*S

// ---------------------------------------------------------------- converts
__global__ __launch_bounds__(256) void cvt_f32_bf16(const float* __restrict__ in,
                                                    unsigned short* __restrict__ out,
                                                    int n) {
    int i = blockIdx.x * 256 + threadIdx.x;
    if (i < n) out[i] = f2bf(in[i]);
}

// w: [K][N] fp32  ->  wt: [N][K] bf16
__global__ __launch_bounds__(256) void transpose_cvt(const float* __restrict__ w,
                                                     unsigned short* __restrict__ wt,
                                                     int K, int N) {
    int i = blockIdx.x * 256 + threadIdx.x;
    if (i < K * N) {
        int k = i / N, n = i - k * N;
        wt[(size_t)n * K + k] = f2bf(w[i]);
    }
}

// ---------------------------------------------------------------- GEMM
// C[M,N] = A[M,K](bf16) * Bt[N,K](bf16)^T + bias, with epilogue modes.
// mode 0: outF fp32 row-major
// mode 2: outB bf16 row-major with ReLU
// mode 3: Q/K layout  outB[((b*NH+h)*SEQ+s)*DKH+d]
// mode 4: V  layout   outB[((b*NH+h)*DKH+d)*SEQ+s]
__global__ __launch_bounds__(256) void gemm_bf16(const unsigned short* __restrict__ A,
                                                 const unsigned short* __restrict__ Bt,
                                                 const float* __restrict__ bias,
                                                 float* __restrict__ outF,
                                                 unsigned short* __restrict__ outB,
                                                 int M, int N, int K, int mode) {
    __shared__ __align__(16) unsigned short ldsA[2][128 * 32];
    __shared__ __align__(16) unsigned short ldsB[2][128 * 32];

    const int t = threadIdx.x;
    const int lane = t & 31, w = t >> 5;
    const int lrow = lane & 15, hi = lane >> 4;
    const int wm = w >> 2, wn = w & 3;          // 2 x 4 wave grid
    const int row0 = blockIdx.y * 128, col0 = blockIdx.x * 128;

    // stage one 128x32 A-tile + 128x32 B-tile into buffer bsel (async DMA)
    auto stage = [&](int k0, int bsel) {
#pragma unroll
        for (int i = 0; i < 2; ++i) {
            int idx = t + i * 256;              // 0..511
            int r = idx >> 2, s = idx & 3;
            cp16_async(&ldsA[bsel][r * 32 + s * 8],
                       &A[(size_t)(row0 + r) * K + k0 + s * 8]);
            cp16_async(&ldsB[bsel][r * 32 + s * 8],
                       &Bt[(size_t)(col0 + r) * K + k0 + s * 8]);
        }
    };

    v8f acc[4][2];
#pragma unroll
    for (int ms = 0; ms < 4; ++ms)
#pragma unroll
        for (int ns = 0; ns < 2; ++ns) acc[ms][ns] = vzero8();

    const int nsteps = K >> 5;
    stage(0, 0);

    for (int ks = 0; ks < nsteps; ++ks) {
        ASYNC_WAIT();            // own async copies from previous stage done
        __syncthreads();         // tiles complete; prior buffer readers done
        if (ks + 1 < nsteps) stage((ks + 1) << 5, (ks + 1) & 1);

        const int cur = ks & 1;
        Frag af[4], bf[2];
#pragma unroll
        for (int ms = 0; ms < 4; ++ms) {
            const unsigned short* rp = &ldsA[cur][(wm * 64 + ms * 16 + lrow) * 32];
            af[ms].u4[0] = *(const uint4*)(rp + hi * 8);
            af[ms].u4[1] = *(const uint4*)(rp + 16 + hi * 8);
        }
#pragma unroll
        for (int ns = 0; ns < 2; ++ns) {
            const unsigned short* rp = &ldsB[cur][(wn * 32 + ns * 16 + lrow) * 32];
            bf[ns].u4[0] = *(const uint4*)(rp + hi * 8);
            bf[ns].u4[1] = *(const uint4*)(rp + 16 + hi * 8);
        }
#pragma unroll
        for (int ms = 0; ms < 4; ++ms)
#pragma unroll
            for (int ns = 0; ns < 2; ++ns)
                acc[ms][ns] = wmma_bf16(af[ms], bf[ns], acc[ms][ns]);
    }

    // epilogue
#pragma unroll
    for (int ms = 0; ms < 4; ++ms) {
#pragma unroll
        for (int ns = 0; ns < 2; ++ns) {
            int gm = row0 + wm * 64 + ms * 16;
            int gn = col0 + wn * 32 + ns * 16 + lrow;
            float bv = bias ? bias[gn] : 0.0f;
#pragma unroll
            for (int r = 0; r < 8; ++r) {
                int grow = gm + r + 8 * hi;
                float v = acc[ms][ns][r] + bv;
                if (mode == 0) {
                    outF[(size_t)grow * N + gn] = v;
                } else if (mode == 2) {
                    outB[(size_t)grow * N + gn] = f2bf(fmaxf(v, 0.0f));
                } else if (mode == 3) {
                    int hh = gn >> 6, dd = gn & 63;
                    int bb = grow >> 11, ss = grow & 2047;
                    outB[(((size_t)(bb * NH + hh)) * SEQ + ss) * DKH + dd] = f2bf(v);
                } else { // mode 4
                    int hh = gn >> 6, dd = gn & 63;
                    int bb = grow >> 11, ss = grow & 2047;
                    outB[(((size_t)(bb * NH + hh)) * DKH + dd) * SEQ + ss] = f2bf(v);
                }
            }
        }
    }
}

// ---------------------------------------------------------------- attention
// Q,K : [B,NH,SEQ,DKH] bf16 ; Vt : [B,NH,DKH,SEQ] bf16
// ctxb: [B,SEQ,DM] bf16. 128 threads = 4 waves; wave handles 16 query rows.
__global__ __launch_bounds__(128) void attn_kernel(const unsigned short* __restrict__ Q,
                                                   const unsigned short* __restrict__ Kb,
                                                   const unsigned short* __restrict__ Vt,
                                                   const unsigned char* __restrict__ mask,
                                                   unsigned short* __restrict__ ctxb) {
    __shared__ __align__(16) unsigned short ldsK[32 * 64];      // [key][d]
    __shared__ __align__(16) unsigned short ldsVt[64 * 32];     // [d][key]
    __shared__ __align__(16) unsigned short ldsP[4 * 16 * 32];  // per-wave P

    const int bid = blockIdx.x;
    const int bh = bid >> 5;             // 0..23
    const int qb = bid & 31;             // q block of 64
    const int bidx = bh / NH;
    const int hd = bh - bidx * NH;

    const int t = threadIdx.x;
    const int w = t >> 5, lane = t & 31;
    const int lrow = lane & 15, hi = lane >> 4;
    const int q0 = qb * 64 + w * 16;     // first query row of this wave (within S)

    // load Q fragments for the wave's 16 rows (d chunks 0 and 32)
    Frag qf[2];
    const unsigned short* qrow = Q + ((size_t)bh * SEQ + q0 + lrow) * DKH;
#pragma unroll
    for (int i = 0; i < 2; ++i) {
        int d0 = i * 32;
        qf[i].u4[0] = *(const uint4*)(qrow + d0 + hi * 8);
        qf[i].u4[1] = *(const uint4*)(qrow + d0 + 16 + hi * 8);
    }

    v8f O[4];
#pragma unroll
    for (int i = 0; i < 4; ++i) O[i] = vzero8();
    v8f m_i, l_i;
#pragma unroll
    for (int r = 0; r < 8; ++r) { m_i[r] = -3.0e38f; l_i[r] = 0.0f; }

    const unsigned char* mrow = mask + (size_t)bidx * SEQ * SEQ;

    for (int j0 = 0; j0 < SEQ; j0 += 32) {
        __syncthreads();   // all waves done reading ldsK/ldsVt from prev iter
        // stage K block [32 keys][64 d] and Vt block [64 d][32 keys] via DMA
#pragma unroll
        for (int i = 0; i < 2; ++i) {
            int idx = t + i * 128;               // 0..255
            int kr = idx >> 3, sg = idx & 7;
            cp16_async(&ldsK[kr * 64 + sg * 8],
                       &Kb[((size_t)bh * SEQ + j0 + kr) * DKH + sg * 8]);
            int vr = idx >> 2, sv = idx & 3;
            cp16_async(&ldsVt[vr * 32 + sv * 8],
                       &Vt[((size_t)bh * DKH + vr) * SEQ + j0 + sv * 8]);
        }
        ASYNC_WAIT();
        __syncthreads();

        // scores: S[16 q][32 key]
        v8f st[2];
#pragma unroll
        for (int n0i = 0; n0i < 2; ++n0i) {
            v8f a = vzero8();
#pragma unroll
            for (int d0i = 0; d0i < 2; ++d0i) {
                Frag bf;
                const unsigned short* rp = &ldsK[(n0i * 16 + lrow) * 64 + d0i * 32];
                bf.u4[0] = *(const uint4*)(rp + hi * 8);
                bf.u4[1] = *(const uint4*)(rp + 16 + hi * 8);
                a = wmma_bf16(qf[d0i], bf, a);
            }
            st[n0i] = a;
        }
        // scale + mask
#pragma unroll
        for (int n0i = 0; n0i < 2; ++n0i) {
#pragma unroll
            for (int r = 0; r < 8; ++r) {
                int qg = q0 + r + 8 * hi;
                int kg = j0 + n0i * 16 + lrow;
                float s = st[n0i][r] * 0.125f;
                if (mrow[(size_t)qg * SEQ + kg]) s = -1.0e9f;
                st[n0i][r] = s;
            }
        }
        // online softmax
        v8f mnew, alpha;
#pragma unroll
        for (int r = 0; r < 8; ++r) {
            float mx = fmaxf(st[0][r], st[1][r]);
#pragma unroll
            for (int o = 1; o < 16; o <<= 1) mx = fmaxf(mx, __shfl_xor(mx, o, 16));
            mnew[r] = fmaxf(m_i[r], mx);
            alpha[r] = __expf(m_i[r] - mnew[r]);
        }
#pragma unroll
        for (int n0i = 0; n0i < 2; ++n0i)
#pragma unroll
            for (int r = 0; r < 8; ++r)
                st[n0i][r] = __expf(st[n0i][r] - mnew[r]);
#pragma unroll
        for (int r = 0; r < 8; ++r) {
            float sm = st[0][r] + st[1][r];
#pragma unroll
            for (int o = 1; o < 16; o <<= 1) sm += __shfl_xor(sm, o, 16);
            l_i[r] = l_i[r] * alpha[r] + sm;
            m_i[r] = mnew[r];
        }
#pragma unroll
        for (int i = 0; i < 4; ++i)
#pragma unroll
            for (int r = 0; r < 8; ++r) O[i][r] *= alpha[r];

        // stage P (D-layout -> LDS) and reload as A fragment
        unsigned short* pw = &ldsP[w * 16 * 32];
#pragma unroll
        for (int n0i = 0; n0i < 2; ++n0i)
#pragma unroll
            for (int r = 0; r < 8; ++r)
                pw[(r + 8 * hi) * 32 + n0i * 16 + lrow] = f2bf(st[n0i][r]);
        __syncthreads();

        Frag pf;
        {
            const unsigned short* rp = pw + lrow * 32;
            pf.u4[0] = *(const uint4*)(rp + hi * 8);
            pf.u4[1] = *(const uint4*)(rp + 16 + hi * 8);
        }
        // O += P[16x32] * V[32x64]
#pragma unroll
        for (int n0 = 0; n0 < 4; ++n0) {
            Frag bf;
            const unsigned short* vp = &ldsVt[(n0 * 16 + lrow) * 32];
            bf.u4[0] = *(const uint4*)(vp + hi * 8);
            bf.u4[1] = *(const uint4*)(vp + 16 + hi * 8);
            O[n0] = wmma_bf16(pf, bf, O[n0]);
        }
    }

    // epilogue: ctx = O / l, scattered to [B,S,DM] bf16
    v8f inv;
#pragma unroll
    for (int r = 0; r < 8; ++r) inv[r] = 1.0f / l_i[r];
#pragma unroll
    for (int n0 = 0; n0 < 4; ++n0) {
#pragma unroll
        for (int r = 0; r < 8; ++r) {
            int grow = bidx * SEQ + q0 + r + 8 * hi;
            int gcol = hd * DKH + n0 * 16 + lrow;
            ctxb[(size_t)grow * DM + gcol] = f2bf(O[n0][r] * inv[r]);
        }
    }
}

// ---------------------------------------------------------------- LN + residual
// y[row] = xres[row] + LayerNorm(a[row]) * g + b ; optional bf16 mirror
__global__ __launch_bounds__(256) void ln_add_kernel(const float* __restrict__ xres,
                                                     const float* __restrict__ a,
                                                     const float* __restrict__ g,
                                                     const float* __restrict__ beta,
                                                     float* __restrict__ y,
                                                     unsigned short* __restrict__ yb) {
    __shared__ float red[256];
    const int row = blockIdx.x;
    const int t = threadIdx.x;
    const size_t base = (size_t)row * DM;

    float v0 = a[base + t], v1 = a[base + t + 256], v2 = a[base + t + 512];
    red[t] = v0 + v1 + v2;
    __syncthreads();
    for (int o = 128; o > 0; o >>= 1) {
        if (t < o) red[t] += red[t + o];
        __syncthreads();
    }
    float mu = red[0] * (1.0f / DM);
    __syncthreads();

    float d0 = v0 - mu, d1 = v1 - mu, d2 = v2 - mu;
    red[t] = d0 * d0 + d1 * d1 + d2 * d2;
    __syncthreads();
    for (int o = 128; o > 0; o >>= 1) {
        if (t < o) red[t] += red[t + o];
        __syncthreads();
    }
    float rstd = rsqrtf(red[0] * (1.0f / DM) + 1.0e-5f);

#pragma unroll
    for (int i = 0; i < 3; ++i) {
        int c = t + i * 256;
        float dv = (i == 0 ? d0 : (i == 1 ? d1 : d2));
        float val = xres[base + c] + dv * rstd * g[c] + beta[c];
        y[base + c] = val;
        if (yb) yb[base + c] = f2bf(val);
    }
}

// ---------------------------------------------------------------- launcher
extern "C" void kernel_launch(void* const* d_in, const int* in_sizes, int n_in,
                              void* d_out, int out_size, void* d_ws, size_t ws_size,
                              hipStream_t stream) {
    (void)in_sizes; (void)n_in; (void)out_size; (void)ws_size;

    const float* x    = (const float*)d_in[0];
    const unsigned char* mask = (const unsigned char*)d_in[1];
    const float* wq = (const float*)d_in[2];
    const float* bq = (const float*)d_in[3];
    const float* wk = (const float*)d_in[4];
    const float* bk = (const float*)d_in[5];
    const float* wv = (const float*)d_in[6];
    const float* bv = (const float*)d_in[7];
    const float* wo = (const float*)d_in[8];
    const float* bo = (const float*)d_in[9];
    const float* w1 = (const float*)d_in[10];
    const float* b1 = (const float*)d_in[11];
    const float* w2 = (const float*)d_in[12];
    const float* b2 = (const float*)d_in[13];
    const float* ln_g = (const float*)d_in[14];
    const float* ln_b = (const float*)d_in[15];
    float* out = (float*)d_out;

    size_t off = 0;
    char* base = (char*)d_ws;
    auto take = [&](size_t bytes) -> void* {
        void* r = base + off;
        off += (bytes + 255) & ~(size_t)255;
        return r;
    };
    unsigned short* wqT = (unsigned short*)take((size_t)DM * DM * 2);
    unsigned short* wkT = (unsigned short*)take((size_t)DM * DM * 2);
    unsigned short* wvT = (unsigned short*)take((size_t)DM * DM * 2);
    unsigned short* woT = (unsigned short*)take((size_t)DM * DM * 2);
    unsigned short* w1T = (unsigned short*)take((size_t)DM * HID * 2);
    unsigned short* w2T = (unsigned short*)take((size_t)HID * DM * 2);
    unsigned short* xb  = (unsigned short*)take((size_t)ROWS * DM * 2);
    unsigned short* Qb  = (unsigned short*)take((size_t)ROWS * DM * 2);
    unsigned short* Kb  = (unsigned short*)take((size_t)ROWS * DM * 2);
    unsigned short* Vtb = (unsigned short*)take((size_t)ROWS * DM * 2);
    unsigned short* ctxb= (unsigned short*)take((size_t)ROWS * DM * 2);
    unsigned short* ybB = (unsigned short*)take((size_t)ROWS * DM * 2);
    unsigned short* hB  = (unsigned short*)take((size_t)ROWS * HID * 2);
    float* aF = (float*)take((size_t)ROWS * DM * 4);
    float* yF = (float*)take((size_t)ROWS * DM * 4);

    // 1. bf16 converts
    cvt_f32_bf16<<<(ROWS * DM) / 256, 256, 0, stream>>>(x, xb, ROWS * DM);
    transpose_cvt<<<(DM * DM) / 256, 256, 0, stream>>>(wq, wqT, DM, DM);
    transpose_cvt<<<(DM * DM) / 256, 256, 0, stream>>>(wk, wkT, DM, DM);
    transpose_cvt<<<(DM * DM) / 256, 256, 0, stream>>>(wv, wvT, DM, DM);
    transpose_cvt<<<(DM * DM) / 256, 256, 0, stream>>>(wo, woT, DM, DM);
    transpose_cvt<<<(DM * HID) / 256, 256, 0, stream>>>(w1, w1T, DM, HID);
    transpose_cvt<<<(HID * DM) / 256, 256, 0, stream>>>(w2, w2T, HID, DM);

    // 2. QKV projections (WMMA) with head-layout scatter epilogues
    dim3 g768(DM / 128, ROWS / 128);
    gemm_bf16<<<g768, 256, 0, stream>>>(xb, wqT, bq, nullptr, Qb,  ROWS, DM, DM, 3);
    gemm_bf16<<<g768, 256, 0, stream>>>(xb, wkT, bk, nullptr, Kb,  ROWS, DM, DM, 3);
    gemm_bf16<<<g768, 256, 0, stream>>>(xb, wvT, bv, nullptr, Vtb, ROWS, DM, DM, 4);

    // 3. flash attention (WMMA QK^T and PV)
    attn_kernel<<<(2 * NH) * (SEQ / 64), 128, 0, stream>>>(Qb, Kb, Vtb, mask, ctxb);

    // 4. output projection -> a (fp32)
    gemm_bf16<<<g768, 256, 0, stream>>>(ctxb, woT, bo, aF, nullptr, ROWS, DM, DM, 0);

    // 5. y = x + LN(a); mirror bf16 for FFN
    ln_add_kernel<<<ROWS, 256, 0, stream>>>(x, aF, ln_g, ln_b, yF, ybB);

    // 6. FFN: h = relu(y @ w1 + b1)  (bf16 out), f = h @ w2 + b2 (fp32)
    dim3 g3072(HID / 128, ROWS / 128);
    gemm_bf16<<<g3072, 256, 0, stream>>>(ybB, w1T, b1, nullptr, hB, ROWS, HID, DM, 2);
    gemm_bf16<<<g768, 256, 0, stream>>>(hB, w2T, b2, aF, nullptr, ROWS, DM, HID, 0);

    // 7. z = y + LN(f) -> out
    ln_add_kernel<<<ROWS, 256, 0, stream>>>(yF, aF, ln_g, ln_b, out, nullptr);
}
